// PointNetPlusSegment_13469017440261
// MI455X (gfx1250) — matrix operand order repaired and checked by reading, not compiled, not run on hardware
//
#include <hip/hip_runtime.h>
#include <cstddef>
#include <cstdint>

// ---------------------------------------------------------------------------
// PointNet++ part segmentation forward, CDNA5 (gfx1250, wave32, WMMA + TDM).
// All MLP (1x1 conv) layers run as f16 WMMA GEMMs with f32 accumulation.
// Conv biases are mathematically dropped (they cancel inside batch-norm).
// A-tiles are staged to LDS by the Tensor Data Mover (when the toolchain
// exposes __builtin_amdgcn_tensor_load_to_lds); B-tiles are transpose-packed
// so WMMA fragments are pure ds_load_b128.
// ---------------------------------------------------------------------------

typedef __attribute__((ext_vector_type(16))) _Float16 v16h;
typedef __attribute__((ext_vector_type(8)))  _Float16 v8h;
typedef __attribute__((ext_vector_type(8)))  float    v8f;
typedef __attribute__((ext_vector_type(4)))  unsigned int u32x4;
typedef __attribute__((ext_vector_type(8)))  int      i32x8;
typedef __attribute__((ext_vector_type(4)))  int      i32x4;

#define NBATCH 8
#define NPTS   4096
#define NCLASS 50
#define NCAT   16

#if defined(__gfx1250__) && __has_builtin(__builtin_amdgcn_tensor_load_to_lds)
#define HAVE_TDM 1
#else
#define HAVE_TDM 0
#endif

#if defined(__gfx1250__) && __has_builtin(__builtin_amdgcn_s_wait_tensorcnt)
#define WAIT_TENSORCNT0() __builtin_amdgcn_s_wait_tensorcnt(0)
#else
#define WAIT_TENSORCNT0() asm volatile("s_wait_tensorcnt 0" ::: "memory")
#endif

// ============================ WMMA GEMM ====================================
// Y[cout][Ntot] = W[cout][cin] * X[cin][Ntot], f16 in, f32 out.
// Block = 128 threads = 4 waves. Block tile: 16 rows x 256 cols.
// Each wave owns a 16x64 strip = 4 accumulators; K stepped by 32 via LDS.
// LDS layout is K-major for both tiles so every WMMA fragment is two
// contiguous 16B runs per lane (ds_load_b128).
__global__ __launch_bounds__(128)
void wmma_gemm_f16_kernel(const _Float16* __restrict__ A,   // [cout][cin]
                          const _Float16* __restrict__ Bm,  // [cin][Ntot]
                          float* __restrict__ C,            // [cout][Ntot]
                          int cout, int cin, int Ntot)
{
    __shared__ __align__(16) _Float16 sAt[16][32];   // [m][k], K-contiguous
    __shared__ __align__(16) _Float16 sBt[256][40];  // [col][k] (+pad, 80B row)

    const int tid  = threadIdx.x;
    const int wave = tid >> 5;
    const int lane = tid & 31;
    const int half = lane >> 4;   // 0: lanes 0-15, 1: lanes 16-31
    const int l16  = lane & 15;

    const int rowBase = blockIdx.y * 16;
    const int colBase = blockIdx.x * 256;

    v8f acc[4];
    #pragma unroll
    for (int i = 0; i < 4; ++i)
        acc[i] = (v8f){0.f, 0.f, 0.f, 0.f, 0.f, 0.f, 0.f, 0.f};

    for (int k0 = 0; k0 < cin; k0 += 32) {
#if HAVE_TDM
        // --- A tile via Tensor Data Mover: 2D tile, dim0=K (contiguous),
        //     dim1=M (stride cin). Remaining-extent dims give free OOB zero
        //     padding at the cin/cout edges.
        if (wave == 0) {
            unsigned long long gaddr =
                (unsigned long long)A + 2ull * ((unsigned long long)rowBase * cin + k0);
            unsigned int ldsA = (unsigned int)(unsigned long long)(uintptr_t)&sAt[0][0];
            unsigned int d0 = (unsigned int)(cin - k0);       // remaining K
            unsigned int d1 = (unsigned int)(cout - rowBase); // remaining M
            u32x4 g0;
            g0[0] = 1u;                                   // count=1, user mode
            g0[1] = ldsA;                                 // lds_addr (bytes)
            g0[2] = (unsigned int)gaddr;                  // global_addr[31:0]
            g0[3] = (unsigned int)(gaddr >> 32) | 0x80000000u; // addr[56:32] | type=2
            i32x8 g1;
            g1[0] = 0x00010000;                           // data_size=2B, mask=0
            g1[1] = (int)((d0 & 0xFFFFu) << 16);          // tensor_dim0[15:0]
            g1[2] = (int)((d0 >> 16) | ((d1 & 0xFFFFu) << 16)); // dim0 hi | dim1 lo
            g1[3] = (int)((d1 >> 16) | (32u << 16));      // dim1 hi | tile_dim0=32
            g1[4] = 16;                                   // tile_dim1=16
            g1[5] = cin;                                  // tensor_dim0_stride lo
            g1[6] = 0;                                    // stride hi | dim1_stride lo
            g1[7] = 0;
            i32x4 z4 = {0, 0, 0, 0};
#if __clang_major__ >= 23
            i32x8 z8 = {0, 0, 0, 0, 0, 0, 0, 0};
            __builtin_amdgcn_tensor_load_to_lds(g0, g1, z4, z4, z8, 0);
#else
            __builtin_amdgcn_tensor_load_to_lds(g0, g1, z4, z4, 0);
#endif
        }
#else
        // --- Fallback: manual A staging (elementwise, zero padded).
        for (int i = tid; i < 16 * 32; i += 128) {
            int r = i >> 5, k = i & 31;
            int gr = rowBase + r, gk = k0 + k;
            sAt[r][k] = (gr < cout && gk < cin) ? A[(size_t)gr * cin + gk]
                                                : (_Float16)0.f;
        }
#endif
        // --- B tile transpose-pack: each task packs 8 strided K values of one
        //     column (coalesced across threads) into one ds_store_b128.
        //     Uniform fast path when the whole K-slab is in-bounds (the common
        //     case) avoids per-element exec-mask churn.
        const bool fullK = (k0 + 32 <= cin);
        for (int task = tid; task < 256 * 4; task += 128) {
            int col = task & 255, kg = task >> 8;
            int gc = colBase + col;
            v8h t;
            if (fullK && gc < Ntot) {
                const _Float16* src = Bm + (size_t)(k0 + kg * 8) * Ntot + gc;
                #pragma unroll
                for (int j = 0; j < 8; ++j)
                    t[j] = src[(size_t)j * Ntot];
            } else {
                #pragma unroll
                for (int j = 0; j < 8; ++j) {
                    int gk = k0 + kg * 8 + j;
                    t[j] = (gk < cin && gc < Ntot) ? Bm[(size_t)gk * Ntot + gc]
                                                   : (_Float16)0.f;
                }
            }
            *(v8h*)&sBt[col][kg * 8] = t;
        }
        if (k0 + 32 < cin)  // global_prefetch_b8 for the next K slab
            __builtin_prefetch(&Bm[(size_t)(k0 + 32) * Ntot + colBase + tid], 0, 1);
#if HAVE_TDM
        if (wave == 0) WAIT_TENSORCNT0();
#endif
        __syncthreads();

        // A fragment: lane half=0 -> K {0..7,16..23}, half=1 -> K {8..15,24..31}
        v8h alo = *(const v8h*)&sAt[l16][half * 8];
        v8h ahi = *(const v8h*)&sAt[l16][16 + half * 8];
        v16h afrag = __builtin_shufflevector(alo, ahi,
            0, 1, 2, 3, 4, 5, 6, 7, 8, 9, 10, 11, 12, 13, 14, 15);

        #pragma unroll
        for (int nt = 0; nt < 4; ++nt) {
            int col = wave * 64 + nt * 16 + l16;
            v8h blo = *(const v8h*)&sBt[col][half * 8];
            v8h bhi = *(const v8h*)&sBt[col][16 + half * 8];
            v16h bfrag = __builtin_shufflevector(blo, bhi,
                0, 1, 2, 3, 4, 5, 6, 7, 8, 9, 10, 11, 12, 13, 14, 15);
            acc[nt] = __builtin_amdgcn_wmma_f32_16x16x32_f16(
                false, afrag, false, bfrag, (short)0, acc[nt], false, false);
        }
        __syncthreads();
    }

    // C/D layout: VGPR r -> M = half*8 + r, N = lane&15 within each tile.
    #pragma unroll
    for (int nt = 0; nt < 4; ++nt) {
        #pragma unroll
        for (int r = 0; r < 8; ++r) {
            int row = rowBase + half * 8 + r;
            int col = colBase + wave * 64 + nt * 16 + l16;
            if (row < cout && col < Ntot)
                C[(size_t)row * Ntot + col] = acc[nt][r];
        }
    }
}

// ============================ BatchNorm ====================================
__global__ void bn_stats_kernel(const float* __restrict__ Y, int Ntot,
                                float* __restrict__ mu, float* __restrict__ rsig)
{
    int c = blockIdx.x;
    const float* row = Y + (size_t)c * Ntot;
    float s = 0.f, s2 = 0.f;
    for (int i = threadIdx.x; i < Ntot; i += blockDim.x) {
        float v = row[i];
        s += v; s2 += v * v;
    }
    __shared__ float ss[256], ss2[256];
    ss[threadIdx.x] = s; ss2[threadIdx.x] = s2;
    __syncthreads();
    for (int st = 128; st > 0; st >>= 1) {
        if (threadIdx.x < st) {
            ss[threadIdx.x]  += ss[threadIdx.x + st];
            ss2[threadIdx.x] += ss2[threadIdx.x + st];
        }
        __syncthreads();
    }
    if (threadIdx.x == 0) {
        float m = ss[0] / (float)Ntot;
        float v = ss2[0] / (float)Ntot - m * m;
        mu[c]   = m;
        rsig[c] = rsqrtf(v + 1e-5f);
    }
}

__global__ void bn_apply_relu_kernel(const float* __restrict__ Y,
                                     const float* __restrict__ mu,
                                     const float* __restrict__ rsig,
                                     const float* __restrict__ g,
                                     const float* __restrict__ beta,
                                     _Float16* __restrict__ X,
                                     int cout, int Ntot)
{
    long i = (long)blockIdx.x * blockDim.x + threadIdx.x;
    long tot = (long)cout * Ntot;
    if (i >= tot) return;
    int c = (int)(i / Ntot);
    float v = (Y[i] - mu[c]) * rsig[c] * g[c] + beta[c];
    X[i] = (_Float16)fmaxf(v, 0.f);
}

// ============================ Geometry =====================================
__global__ void cast_f32_f16_kernel(const float* __restrict__ s,
                                    _Float16* __restrict__ d, long n)
{
    long i = (long)blockIdx.x * blockDim.x + threadIdx.x;
    if (i < n) d[i] = (_Float16)s[i];
}

// pc [B][3][N] -> xyz [B][N][3]
__global__ void transpose_pc_kernel(const float* __restrict__ pc,
                                    float* __restrict__ xyz)
{
    int t = blockIdx.x * blockDim.x + threadIdx.x;
    if (t >= NBATCH * NPTS) return;
    int n = t % NPTS, b = t / NPTS;
    for (int c = 0; c < 3; ++c)
        xyz[(size_t)t * 3 + c] = pc[((size_t)b * 3 + c) * NPTS + n];
}

// Farthest point sampling. One block per batch. xyz [B][N][3], out [B][M][3].
__global__ void fps_kernel(const float* __restrict__ xyz, int N, int M,
                           float* __restrict__ new_xyz)
{
    int b = blockIdx.x;
    const float* x = xyz + (size_t)b * N * 3;
    __shared__ float dist[4096];
    __shared__ float rv[256];
    __shared__ int   ri[256];
    __shared__ int   cur;
    for (int i = threadIdx.x; i < N; i += 256) dist[i] = 1e10f;
    if (threadIdx.x == 0) cur = 0;
    __syncthreads();
    for (int it = 0; it < M; ++it) {
        int c = cur;
        float px = x[c * 3], py = x[c * 3 + 1], pz = x[c * 3 + 2];
        if (threadIdx.x == 0) {
            new_xyz[((size_t)b * M + it) * 3 + 0] = px;
            new_xyz[((size_t)b * M + it) * 3 + 1] = py;
            new_xyz[((size_t)b * M + it) * 3 + 2] = pz;
        }
        float best = -1.f; int bi = 0;
        for (int n = threadIdx.x; n < N; n += 256) {
            float dx = x[n * 3] - px, dy = x[n * 3 + 1] - py, dz = x[n * 3 + 2] - pz;
            float d = dx * dx + dy * dy + dz * dz;
            float dm = fminf(dist[n], d);
            dist[n] = dm;
            if (dm > best) { best = dm; bi = n; }
        }
        rv[threadIdx.x] = best; ri[threadIdx.x] = bi;
        __syncthreads();
        for (int st = 128; st > 0; st >>= 1) {
            if (threadIdx.x < st && rv[threadIdx.x + st] > rv[threadIdx.x]) {
                rv[threadIdx.x] = rv[threadIdx.x + st];
                ri[threadIdx.x] = ri[threadIdx.x + st];
            }
            __syncthreads();
        }
        if (threadIdx.x == 0) cur = ri[0];
        __syncthreads();
    }
}

// Ball query: first K in-radius indices in scan order, padded with first hit.
__global__ void ball_query_kernel(const float* __restrict__ xyz,
                                  const float* __restrict__ new_xyz,
                                  int N, int M, float r2, int K,
                                  int* __restrict__ idx)
{
    int t = blockIdx.x * blockDim.x + threadIdx.x;
    if (t >= NBATCH * M) return;
    int b = t / M;
    const float* x = xyz + (size_t)b * N * 3;
    const float* q = new_xyz + (size_t)t * 3;
    int* out = idx + (size_t)t * K;
    float qx = q[0], qy = q[1], qz = q[2];
    int cnt = 0, first = 0; bool found = false;
    for (int n = 0; n < N && cnt < K; ++n) {
        float dx = x[n * 3] - qx, dy = x[n * 3 + 1] - qy, dz = x[n * 3 + 2] - qz;
        float d = dx * dx + dy * dy + dz * dz;
        if (d < r2) {
            if (!found) { found = true; first = n; }
            out[cnt++] = n;
        }
    }
    for (; cnt < K; ++cnt) out[cnt] = first;
}

// Grouped centered coordinates -> GEMM X rows 0..2 (f16).
__global__ void group_xyz_kernel(const float* __restrict__ pts,     // [B][N][3]
                                 const float* __restrict__ new_xyz, // [B][M][3]
                                 const int* __restrict__ idx,
                                 int N, int M, int K,
                                 _Float16* __restrict__ X, int Ntot)
{
    int t = blockIdx.x * blockDim.x + threadIdx.x;
    if (t >= NBATCH * M * K) return;
    int m = (t / K) % M, b = t / (K * M);
    int n = idx[t];
    for (int c = 0; c < 3; ++c) {
        float v = pts[((size_t)b * N + n) * 3 + c] -
                  new_xyz[((size_t)b * M + m) * 3 + c];
        X[(size_t)c * Ntot + t] = (_Float16)v;
    }
}

// Grouped features -> GEMM X rows rowOff.. (f16). F is [C][B*Mprev].
__global__ void group_feat_kernel(const _Float16* __restrict__ F,
                                  const int* __restrict__ idx,
                                  int C, int Mprev, int M, int K,
                                  _Float16* __restrict__ X, int Ntot, int rowOff)
{
    int t = blockIdx.x * blockDim.x + threadIdx.x;
    if (t >= NBATCH * M * K) return;
    int b = t / (K * M);
    int n = idx[t];
    const size_t fcols = (size_t)NBATCH * Mprev;
    for (int c = 0; c < C; ++c)
        X[(size_t)(rowOff + c) * Ntot + t] = F[(size_t)c * fcols + (size_t)b * Mprev + n];
}

// Max over neighbors: Xin [C][B*M*K] -> Fout[rowOff+c][b*M+m].
__global__ void max_over_k_kernel(const _Float16* __restrict__ Xin,
                                  int C, int M, int K,
                                  _Float16* __restrict__ Fout, int rowOff, int FoutCols)
{
    int t = blockIdx.x * blockDim.x + threadIdx.x;
    if (t >= NBATCH * M * C) return;
    int c = t % C, m = (t / C) % M, b = t / (C * M);
    int Ntot = NBATCH * M * K;
    const _Float16* row = Xin + (size_t)c * Ntot + ((size_t)b * M + m) * K;
    float best = -1e30f;
    for (int k = 0; k < K; ++k) best = fmaxf(best, (float)row[k]);
    Fout[(size_t)(rowOff + c) * FoutCols + (size_t)b * M + m] = (_Float16)best;
}

// Per-batch global max over 128 columns of G [1024][B*128] -> gmax[c][b] f32.
__global__ void global_max_kernel(const _Float16* __restrict__ G,
                                  float* __restrict__ gmax)
{
    int t = blockIdx.x * blockDim.x + threadIdx.x;
    if (t >= 1024 * NBATCH) return;
    int b = t % NBATCH, c = t / NBATCH;
    const _Float16* row = G + (size_t)c * (NBATCH * 128) + b * 128;
    float best = -1e30f;
    for (int m = 0; m < 128; ++m) best = fmaxf(best, (float)row[m]);
    gmax[(size_t)c * NBATCH + b] = best;
}

// X[0..639] = f_sa2, X[640..1663] = broadcast gmax. Cols = B*128.
__global__ void concat_local_kernel(const _Float16* __restrict__ fsa2,
                                    const float* __restrict__ gmax,
                                    _Float16* __restrict__ X)
{
    int t = blockIdx.x * blockDim.x + threadIdx.x;
    const int cols = NBATCH * 128;
    if (t >= 1664 * cols) return;
    int col = t % cols, c = t / cols, b = col / 128;
    X[t] = (c < 640) ? fsa2[(size_t)c * cols + col]
                     : (_Float16)gmax[(size_t)(c - 640) * NBATCH + b];
}

// 3-NN inverse-distance interpolation into X rows [rowOff..rowOff+C).
__global__ void interpolate_kernel(const _Float16* __restrict__ F, int C,
                                   const float* __restrict__ upc,  // [B][Nu][3]
                                   const float* __restrict__ kpc,  // [B][Mk][3]
                                   int Nu, int Mk,
                                   _Float16* __restrict__ X, int Ntot, int rowOff)
{
    int t = blockIdx.x * blockDim.x + threadIdx.x;
    if (t >= NBATCH * Nu) return;
    int b = t / Nu;
    const float* u  = upc + (size_t)t * 3;
    const float* kp = kpc + (size_t)b * Mk * 3;
    float ux = u[0], uy = u[1], uz = u[2];
    float d0 = 1e30f, d1 = 1e30f, d2 = 1e30f;
    int i0 = 0, i1 = 0, i2 = 0;
    for (int m = 0; m < Mk; ++m) {
        float dx = kp[m * 3] - ux, dy = kp[m * 3 + 1] - uy, dz = kp[m * 3 + 2] - uz;
        float d = dx * dx + dy * dy + dz * dz;
        if (d < d0)      { d2 = d1; i2 = i1; d1 = d0; i1 = i0; d0 = d; i0 = m; }
        else if (d < d1) { d2 = d1; i2 = i1; d1 = d;  i1 = m; }
        else if (d < d2) { d2 = d;  i2 = m; }
    }
    float w0 = 1.f / (d0 + 1e-8f), w1 = 1.f / (d1 + 1e-8f), w2 = 1.f / (d2 + 1e-8f);
    float ws = w0 + w1 + w2;
    w0 /= ws; w1 /= ws; w2 /= ws;
    const size_t fcols = (size_t)NBATCH * Mk;
    for (int c = 0; c < C; ++c) {
        const _Float16* fr = F + (size_t)c * fcols + (size_t)b * Mk;
        float v = w0 * (float)fr[i0] + w1 * (float)fr[i1] + w2 * (float)fr[i2];
        X[(size_t)(rowOff + c) * Ntot + t] = (_Float16)v;
    }
}

// Generic contiguous row copy into a larger X at row offset (same #cols).
__global__ void copy_rows_kernel(const _Float16* __restrict__ src,
                                 _Float16* __restrict__ dst,
                                 long total, long dstOff)
{
    long t = (long)blockIdx.x * blockDim.x + threadIdx.x;
    if (t < total) dst[dstOff + t] = src[t];
}

// fp2 extra rows: 256..258 = raw xyz, 259..274 = one-hot(label).
__global__ void fp2_extra_kernel(const float* __restrict__ pc,
                                 const int* __restrict__ labels,
                                 _Float16* __restrict__ X, int Ntot)
{
    int t = blockIdx.x * blockDim.x + threadIdx.x;
    if (t >= NBATCH * NPTS) return;
    int n = t % NPTS, b = t / NPTS;
    for (int c = 0; c < 3; ++c)
        X[(size_t)(256 + c) * Ntot + t] = (_Float16)pc[((size_t)b * 3 + c) * NPTS + n];
    int lab = labels[b];
    for (int c = 0; c < NCAT; ++c)
        X[(size_t)(259 + c) * Ntot + t] = (_Float16)((c == lab) ? 1.f : 0.f);
}

// Final permute [o][b*N+n] -> [b][o][n] with bias.
__global__ void seg_final_kernel(const float* __restrict__ Y,
                                 const float* __restrict__ bias,
                                 float* __restrict__ out)
{
    int t = blockIdx.x * blockDim.x + threadIdx.x;
    if (t >= NBATCH * NCLASS * NPTS) return;
    int n = t % NPTS, o = (t / NPTS) % NCLASS, b = t / (NPTS * NCLASS);
    out[t] = Y[(size_t)o * (NBATCH * NPTS) + (size_t)b * NPTS + n] + bias[o];
}

// ============================ Host orchestration ===========================
struct LayerP {
    const float *W, *b, *beta, *g;
    int cin, cout;
    _Float16* Wh;
};

extern "C" void kernel_launch(void* const* d_in, const int* in_sizes, int n_in,
                              void* d_out, int out_size, void* d_ws, size_t ws_size,
                              hipStream_t stream)
{
    (void)in_sizes; (void)n_in; (void)out_size; (void)ws_size;
    const float* pc     = (const float*)d_in[0];
    const int*   labels = (const int*)d_in[1];

    // params leaves in JAX pytree order (dict keys sorted):
    // fp1, fp2, glob, local, sa1, sa2, seg1, seg_out; each layer {W,b,beta,g}.
    int pi = 2;
    auto grab = [&](int cin, int cout) -> LayerP {
        LayerP L;
        L.W = (const float*)d_in[pi + 0];
        L.b = (const float*)d_in[pi + 1];
        L.beta = (const float*)d_in[pi + 2];
        L.g = (const float*)d_in[pi + 3];
        pi += 4; L.cin = cin; L.cout = cout; L.Wh = nullptr;
        return L;
    };
    LayerP fp1L[2]   = { grab(576, 256), grab(256, 256) };
    LayerP fp2L[2]   = { grab(275, 128), grab(128, 128) };
    LayerP globL[3]  = { grab(640, 256), grab(256, 512), grab(512, 1024) };
    LayerP localL[2] = { grab(1664, 512), grab(512, 256) };
    LayerP sa1L[3][3] = {
        { grab(3, 32), grab(32, 32), grab(32, 64) },
        { grab(3, 64), grab(64, 64), grab(64, 128) },
        { grab(3, 64), grab(64, 96), grab(96, 128) } };
    LayerP sa2L[3][3] = {
        { grab(323, 64),  grab(64, 64),   grab(64, 128) },
        { grab(323, 128), grab(128, 128), grab(128, 256) },
        { grab(323, 128), grab(128, 128), grab(128, 256) } };
    LayerP seg1L = grab(128, 128);
    const float* segW = (const float*)d_in[pi++];
    const float* segb = (const float*)d_in[pi++];

    // ---- workspace bump allocator ----
    char* wsp = (char*)d_ws;
    size_t off = 0;
    auto alloc = [&](size_t bytes) -> char* {
        char* p = wsp + off;
        off = (off + bytes + 255) & ~(size_t)255;
        return p;
    };

    float*    xyz0  = (float*)alloc((size_t)NBATCH * NPTS * 3 * 4);
    float*    nxyz1 = (float*)alloc((size_t)NBATCH * 512 * 3 * 4);
    float*    nxyz2 = (float*)alloc((size_t)NBATCH * 128 * 3 * 4);
    int*      idxb  = (int*)alloc((size_t)NBATCH * 512 * 128 * 4);
    _Float16* fsa1  = (_Float16*)alloc((size_t)320 * NBATCH * 512 * 2);
    _Float16* fsa2  = (_Float16*)alloc((size_t)640 * NBATCH * 128 * 2);
    float*    gmax  = (float*)alloc((size_t)1024 * NBATCH * 4);
    _Float16* locF  = (_Float16*)alloc((size_t)256 * NBATCH * 128 * 2);
    _Float16* Xfp1  = (_Float16*)alloc((size_t)576 * NBATCH * 512 * 2);
    _Float16* ffp1  = (_Float16*)alloc((size_t)256 * NBATCH * 512 * 2);
    _Float16* Xfp2  = (_Float16*)alloc((size_t)275 * NBATCH * NPTS * 2);
    _Float16* ffp2  = (_Float16*)alloc((size_t)128 * NBATCH * NPTS * 2);
    float*    mu    = (float*)alloc(1024 * 4);
    float*    rsig  = (float*)alloc(1024 * 4);

    const size_t maxElems = (size_t)128 * NBATCH * 512 * 128;  // largest activation
    _Float16* Xa = (_Float16*)alloc(maxElems * 2);
    _Float16* Xb = (_Float16*)alloc(maxElems * 2);
    float*    Yb = (float*)alloc(maxElems * 4);

    // ---- convert all weights to f16 once ----
    auto prep = [&](LayerP& L) {
        long n = (long)L.cin * L.cout;
        L.Wh = (_Float16*)alloc((size_t)n * 2);
        cast_f32_f16_kernel<<<(int)((n + 255) / 256), 256, 0, stream>>>(L.W, L.Wh, n);
    };
    for (int i = 0; i < 2; ++i) prep(fp1L[i]);
    for (int i = 0; i < 2; ++i) prep(fp2L[i]);
    for (int i = 0; i < 3; ++i) prep(globL[i]);
    for (int i = 0; i < 2; ++i) prep(localL[i]);
    for (int i = 0; i < 3; ++i) for (int j = 0; j < 3; ++j) prep(sa1L[i][j]);
    for (int i = 0; i < 3; ++i) for (int j = 0; j < 3; ++j) prep(sa2L[i][j]);
    prep(seg1L);
    _Float16* segWh = (_Float16*)alloc((size_t)NCLASS * 128 * 2);
    cast_f32_f16_kernel<<<(NCLASS * 128 + 255) / 256, 256, 0, stream>>>(
        segW, segWh, NCLASS * 128);

    // ---- fused conv+BN+ReLU layer ----
    auto run_layer = [&](const LayerP& L, const _Float16* X, _Float16* Xout, int Ntot) {
        dim3 g((Ntot + 255) / 256, (L.cout + 15) / 16);
        wmma_gemm_f16_kernel<<<g, 128, 0, stream>>>(L.Wh, X, Yb, L.cout, L.cin, Ntot);
        bn_stats_kernel<<<L.cout, 256, 0, stream>>>(Yb, Ntot, mu, rsig);
        long tot = (long)L.cout * Ntot;
        bn_apply_relu_kernel<<<(int)((tot + 255) / 256), 256, 0, stream>>>(
            Yb, mu, rsig, L.g, L.beta, Xout, L.cout, Ntot);
    };

    // ---- geometry prep ----
    transpose_pc_kernel<<<(NBATCH * NPTS + 255) / 256, 256, 0, stream>>>(pc, xyz0);
    fps_kernel<<<NBATCH, 256, 0, stream>>>(xyz0, NPTS, 512, nxyz1);

    // ---- SA1 ----
    {
        const float r[3] = {0.04f, 0.08f, 0.16f};
        const int   K[3] = {32, 64, 128};
        const int coff[3] = {0, 64, 192};
        for (int br = 0; br < 3; ++br) {
            int Kn = K[br], Ntot = NBATCH * 512 * Kn;
            ball_query_kernel<<<(NBATCH * 512 + 255) / 256, 256, 0, stream>>>(
                xyz0, nxyz1, NPTS, 512, r[br] * r[br], Kn, idxb);
            group_xyz_kernel<<<(Ntot + 255) / 256, 256, 0, stream>>>(
                xyz0, nxyz1, idxb, NPTS, 512, Kn, Xa, Ntot);
            run_layer(sa1L[br][0], Xa, Xb, Ntot);
            run_layer(sa1L[br][1], Xb, Xa, Ntot);
            run_layer(sa1L[br][2], Xa, Xb, Ntot);
            int C = sa1L[br][2].cout;
            max_over_k_kernel<<<(NBATCH * 512 * C + 255) / 256, 256, 0, stream>>>(
                Xb, C, 512, Kn, fsa1, coff[br], NBATCH * 512);
        }
    }

    // ---- SA2 ----
    fps_kernel<<<NBATCH, 256, 0, stream>>>(nxyz1, 512, 128, nxyz2);
    {
        const float r[3] = {0.08f, 0.16f, 0.32f};
        const int   K[3] = {16, 32, 64};
        const int coff[3] = {0, 128, 384};
        for (int br = 0; br < 3; ++br) {
            int Kn = K[br], Ntot = NBATCH * 128 * Kn;
            ball_query_kernel<<<(NBATCH * 128 + 255) / 256, 256, 0, stream>>>(
                nxyz1, nxyz2, 512, 128, r[br] * r[br], Kn, idxb);
            group_xyz_kernel<<<(Ntot + 255) / 256, 256, 0, stream>>>(
                nxyz1, nxyz2, idxb, 512, 128, Kn, Xa, Ntot);
            group_feat_kernel<<<(Ntot + 255) / 256, 256, 0, stream>>>(
                fsa1, idxb, 320, 512, 128, Kn, Xa, Ntot, 3);
            run_layer(sa2L[br][0], Xa, Xb, Ntot);
            run_layer(sa2L[br][1], Xb, Xa, Ntot);
            run_layer(sa2L[br][2], Xa, Xb, Ntot);
            int C = sa2L[br][2].cout;
            max_over_k_kernel<<<(NBATCH * 128 * C + 255) / 256, 256, 0, stream>>>(
                Xb, C, 128, Kn, fsa2, coff[br], NBATCH * 128);
        }
    }

    // ---- global feature + local ----
    {
        int cols = NBATCH * 128;
        run_layer(globL[0], fsa2, Xb, cols);
        run_layer(globL[1], Xb, Xa, cols);
        run_layer(globL[2], Xa, Xb, cols);              // Xb: [1024][1024]
        global_max_kernel<<<(1024 * NBATCH + 255) / 256, 256, 0, stream>>>(Xb, gmax);
        concat_local_kernel<<<(1664 * cols + 255) / 256, 256, 0, stream>>>(fsa2, gmax, Xa);
        run_layer(localL[0], Xa, Xb, cols);
        run_layer(localL[1], Xb, locF, cols);           // locF: [256][1024]
    }

    // ---- FP1 ----
    {
        int Ntot = NBATCH * 512;
        interpolate_kernel<<<(Ntot + 255) / 256, 256, 0, stream>>>(
            locF, 256, nxyz1, nxyz2, 512, 128, Xfp1, Ntot, 0);
        long cpy = (long)320 * Ntot;
        copy_rows_kernel<<<(int)((cpy + 255) / 256), 256, 0, stream>>>(
            fsa1, Xfp1, cpy, (long)256 * Ntot);
        run_layer(fp1L[0], Xfp1, Xa, Ntot);
        run_layer(fp1L[1], Xa, ffp1, Ntot);
    }

    // ---- FP2 ----
    {
        int Ntot = NBATCH * NPTS;
        interpolate_kernel<<<(Ntot + 255) / 256, 256, 0, stream>>>(
            ffp1, 256, xyz0, nxyz1, NPTS, 512, Xfp2, Ntot, 0);
        fp2_extra_kernel<<<(Ntot + 255) / 256, 256, 0, stream>>>(pc, labels, Xfp2, Ntot);
        run_layer(fp2L[0], Xfp2, Xa, Ntot);
        run_layer(fp2L[1], Xa, ffp2, Ntot);
    }

    // ---- seg head ----
    {
        int Ntot = NBATCH * NPTS;
        run_layer(seg1L, ffp2, Xa, Ntot);
        dim3 g((Ntot + 255) / 256, (NCLASS + 15) / 16);
        wmma_gemm_f16_kernel<<<g, 128, 0, stream>>>(segWh, Xa, Yb, NCLASS, 128, Ntot);
        seg_final_kernel<<<(NBATCH * NCLASS * NPTS + 255) / 256, 256, 0, stream>>>(
            Yb, segb, (float*)d_out);
    }
}
